// Graph_Inception_12704513261761
// MI455X (gfx1250) — matrix operand order repaired and checked by reading, not compiled, not run on hardware
//
#include <hip/hip_runtime.h>

// ---------------------------------------------------------------------------
// Graph-Inception forward for MI455X (gfx1250, wave32, WMMA).
// All GEMMs (adjacency aggregation + MLPs) run on v_wmma_f32_16x16x32_f16
// with f32 accumulation. Each wave computes a 32x32 output block (2x2 WMMA
// tiles): 16 FLOP/byte from L2 (vs 8 naive) while keeping VGPR pressure
// (~64 core + pipelined loads) low enough to avoid scratch spills.
// Working set stays L2-resident (<180 MB of 192 MB).
// ---------------------------------------------------------------------------

typedef _Float16 half_t;
typedef half_t v16h __attribute__((ext_vector_type(16)));
typedef half_t v8h  __attribute__((ext_vector_type(8)));
typedef float  v8f  __attribute__((ext_vector_type(8)));

#define RROWS 16384      // B*N
#define EPS   1e-5f
#define MT 2             // 16-row tiles per wave
#define NT 2             // 16-col tiles per wave

// ---------------------------------------------------------------------------
// Blocked WMMA GEMM: Y[b] = X[b] (RxK f16 row-major) * W[b]^T (OpxK f16
// row-major, rows >= O zero-padded) + bias. One wave = MT x NT 16x16 tiles.
// A-operand k order used here (contiguous, same for B): lanes 0-15 hold K {0..15},
// lanes 16-31 hold K {16..31} of row M=lane&15  -> one 32B load.
// B-operand k order used here (same as A): lane's 16 halves are K {0..15} (lanes 0-15)
// or {16..31} (lanes 16-31) of column N=lane&15 -> one 32B load of a W row.
// D layout: VGPR r -> row base + (lane>>4)*8 + r, col = base + (lane&15).
// ---------------------------------------------------------------------------
__global__ void __launch_bounds__(256, 1)
gemm_wmma_f16(const half_t* __restrict__ X, long sXb,
              const half_t* __restrict__ Wm, long sWb,
              const float* __restrict__ bias,
              float* __restrict__ Yf, half_t* __restrict__ Yh,
              long sYb, int K, int ldY, int O,
              int mgroups, int ogroups, int totalTiles)
{
    int wave = threadIdx.x >> 5;
    int lane = threadIdx.x & 31;
    int tile = blockIdx.x * (blockDim.x >> 5) + wave;
    if (tile >= totalTiles) return;                 // wave-uniform branch

    int tpb = mgroups * ogroups;
    int b   = tile / tpb;
    int rem = tile - b * tpb;
    int mg  = rem % mgroups;
    int og  = rem / mgroups;

    const half_t* Xb = X + (size_t)b * sXb;
    const half_t* Wb = Wm + (size_t)b * sWb;

    int hi  = lane >> 4;
    int l15 = lane & 15;

    const half_t* xrow[MT];
    const half_t* wrow[NT];
#pragma unroll
    for (int i = 0; i < MT; ++i)
        xrow[i] = Xb + (size_t)((mg * MT + i) * 16 + l15) * K;
#pragma unroll
    for (int j = 0; j < NT; ++j)
        wrow[j] = Wb + (size_t)((og * NT + j) * 16 + l15) * K;

    v8f acc[MT][NT];
#pragma unroll
    for (int i = 0; i < MT; ++i)
#pragma unroll
        for (int j = 0; j < NT; ++j)
            acc[i][j] = (v8f){0.f, 0.f, 0.f, 0.f, 0.f, 0.f, 0.f, 0.f};

    for (int kb = 0; kb < K; kb += 32) {
        v16h av[MT], bv[NT];
#pragma unroll
        for (int i = 0; i < MT; ++i) {
            av[i] = *(const v16h*)(xrow[i] + kb + hi * 16);      // 32B aligned
        }
#pragma unroll
        for (int j = 0; j < NT; ++j)
            bv[j] = *(const v16h*)(wrow[j] + kb + hi * 16);      // 32B aligned
#pragma unroll
        for (int i = 0; i < MT; ++i)
#pragma unroll
            for (int j = 0; j < NT; ++j)
                acc[i][j] = __builtin_amdgcn_wmma_f32_16x16x32_f16(
                    false, av[i], false, bv[j], (short)0, acc[i][j], false, false);
    }

    size_t ybase = (size_t)b * sYb;
#pragma unroll
    for (int j = 0; j < NT; ++j) {
        int ob = (og * NT + j) * 16;
        if (ob >= O) continue;                       // padded sub-tile: skip
        int ocol = ob + l15;
        float badd = bias ? bias[ocol] : 0.0f;
#pragma unroll
        for (int i = 0; i < MT; ++i) {
#pragma unroll
            for (int r = 0; r < 8; ++r) {
                int row = (mg * MT + i) * 16 + hi * 8 + r;
                float v = acc[i][j][r] + badd;
                size_t off = ybase + (size_t)row * ldY + ocol;
                if (Yf) Yf[off] = v;
                if (Yh) Yh[off] = (half_t)v;
            }
        }
    }
}

// ---- per-column mean/var (biased, training-mode BN stats) -----------------
__global__ void col_stats(const float* __restrict__ Y, int Rr, int O,
                          float* __restrict__ mean, float* __restrict__ var)
{
    int col = blockIdx.x;
    float s = 0.f, s2 = 0.f;
    for (int r = threadIdx.x; r < Rr; r += blockDim.x) {
        float v = Y[(size_t)r * O + col];
        s += v; s2 += v * v;
    }
    __shared__ float sh[256], sh2[256];
    sh[threadIdx.x] = s; sh2[threadIdx.x] = s2;
    __syncthreads();
    for (int st = 128; st > 0; st >>= 1) {
        if (threadIdx.x < st) {
            sh[threadIdx.x]  += sh[threadIdx.x + st];
            sh2[threadIdx.x] += sh2[threadIdx.x + st];
        }
        __syncthreads();
    }
    if (threadIdx.x == 0) {
        float mu = sh[0] / (float)Rr;
        mean[col] = mu;
        var[col]  = sh2[0] / (float)Rr - mu * mu;
    }
}

// ---- per-column min (dummy row for neighbor max-pool) ---------------------
__global__ void col_min_k(const float* __restrict__ Y, int Rr, int O,
                          float* __restrict__ mn)
{
    int col = blockIdx.x;
    float m = 3.4e38f;
    for (int r = threadIdx.x; r < Rr; r += blockDim.x)
        m = fminf(m, Y[(size_t)r * O + col]);
    __shared__ float sh[256];
    sh[threadIdx.x] = m;
    __syncthreads();
    for (int st = 128; st > 0; st >>= 1) {
        if (threadIdx.x < st)
            sh[threadIdx.x] = fminf(sh[threadIdx.x], sh[threadIdx.x + st]);
        __syncthreads();
    }
    if (threadIdx.x == 0) mn[col] = sh[0];
}

// ---- bn0 (affine-free) -> h f32 + per-batch transposed f16 copy -----------
__global__ void bn0_apply(const float* __restrict__ x,
                          const float* __restrict__ mean,
                          const float* __restrict__ var,
                          float* __restrict__ h, half_t* __restrict__ HT, int F)
{
    int idx = blockIdx.x * blockDim.x + threadIdx.x;
    if (idx >= RROWS * F) return;
    int r = idx / F, c = idx - r * F;
    float v = (x[idx] - mean[c]) * rsqrtf(var[c] + EPS);
    h[idx] = v;
    int b = r >> 9, n = r & 511;
    HT[((size_t)b * F + c) * 512 + n] = (half_t)v;
}

// ---- relu(adj) -> f16 ------------------------------------------------------
__global__ void adj_relu_f16(const float* __restrict__ adj,
                             half_t* __restrict__ Ah)
{
    int idx = blockIdx.x * blockDim.x + threadIdx.x;
    if (idx < 512 * 512) Ah[idx] = (half_t)fmaxf(adj[idx], 0.f);
}

// ---- weight f32 -> f16, zero-padded rows (O->Op) and cols (K->Kp) ---------
__global__ void cvt_f16_pad(const float* __restrict__ W,
                            half_t* __restrict__ Wh, int O, int Op, int K, int Kp)
{
    int idx = blockIdx.x * blockDim.x + threadIdx.x;
    if (idx >= Op * Kp) return;
    int o = idx / Kp, k = idx - o * Kp;
    Wh[idx] = (half_t)((o < O && k < K) ? W[o * K + k] : 0.f);
}

// ---- neighbor max-pool (dummy index == 16384 selects column-min row) ------
__global__ void maxpool_k(const float* __restrict__ h,
                          const float* __restrict__ dmin,
                          const int* __restrict__ pnl,
                          float* __restrict__ hN, half_t* __restrict__ HTN,
                          int F, int Fn)
{
    int r = blockIdx.x;
    __shared__ int idxs[16];
    if (threadIdx.x < 16) idxs[threadIdx.x] = pnl[(size_t)r * 16 + threadIdx.x];
    __syncthreads();
    int b = r >> 9, n = r & 511;
    for (int f = threadIdx.x; f < F; f += blockDim.x) {
        float m = -3.4e38f;
#pragma unroll
        for (int j = 0; j < 16; ++j) {
            int id = idxs[j];
            float v = (id == RROWS) ? dmin[f] : h[(size_t)id * F + f];
            m = fmaxf(m, v);
        }
        hN[(size_t)r * Fn + f] = m;
        HTN[((size_t)b * Fn + f) * 512 + n] = (half_t)m;
    }
}

// ---- BN + relu -> padded f16 activation buffer (MLP hidden) ---------------
__global__ void bn_relu_f16pad(const float* __restrict__ Y,
                               const float* __restrict__ g,
                               const float* __restrict__ bt,
                               const float* __restrict__ mean,
                               const float* __restrict__ var,
                               half_t* __restrict__ out, int O, int Op)
{
    int idx = blockIdx.x * blockDim.x + threadIdx.x;
    if (idx >= RROWS * Op) return;
    int r = idx / Op, c = idx - r * Op;
    float v = 0.f;
    if (c < O) {
        v = g[c] * (Y[(size_t)r * O + c] - mean[c]) * rsqrtf(var[c] + EPS) + bt[c];
        v = fmaxf(v, 0.f);
    }
    out[idx] = (half_t)v;
}

// ---- BN + relu -> slice of next h (f32) + transposed f16 copy -------------
__global__ void bn_relu_to_h(const float* __restrict__ Y,
                             const float* __restrict__ g,
                             const float* __restrict__ bt,
                             const float* __restrict__ mean,
                             const float* __restrict__ var,
                             float* __restrict__ hN, half_t* __restrict__ HTN,
                             int O, int Fn, int colOff)
{
    int idx = blockIdx.x * blockDim.x + threadIdx.x;
    if (idx >= RROWS * O) return;
    int r = idx / O, c = idx - r * O;
    float v = g[c] * (Y[idx] - mean[c]) * rsqrtf(var[c] + EPS) + bt[c];
    v = fmaxf(v, 0.f);
    hN[(size_t)r * Fn + colOff + c] = v;
    int b = r >> 9, n = r & 511;
    HTN[((size_t)b * Fn + colOff + c) * 512 + n] = (half_t)v;
}

// ---- final pooling: max/argmax, weighted sum, mean over nodes -------------
__global__ void final_pool(const float* __restrict__ h,
                           const float* __restrict__ pw,
                           float* __restrict__ pooled,
                           float* __restrict__ indOut)
{
    int b = blockIdx.x;
    for (int f = threadIdx.x; f < 704; f += blockDim.x) {
        float mx = -3.4e38f; int arg = 0; float sum = 0.f, wsum = 0.f;
        for (int n = 0; n < 512; ++n) {
            float v = h[((size_t)(b * 512 + n)) * 704 + f];
            if (v > mx) { mx = v; arg = n; }       // first-max (jnp.argmax)
            sum  += v;
            wsum += pw[n] * v;
        }
        pooled[b * 2112 + f]        = mx;
        pooled[b * 2112 + 704 + f]  = wsum;
        pooled[b * 2112 + 1408 + f] = sum * (1.0f / 512.0f);
        indOut[b * 704 + f] = (float)arg;
    }
}

// ---- classifier layer 1 + PReLU (fp32, tiny) ------------------------------
__global__ void cls1(const float* __restrict__ pooled,
                     const float* __restrict__ W1, const float* __restrict__ b1,
                     const float* __restrict__ a, float* __restrict__ zp)
{
    int idx = blockIdx.x * blockDim.x + threadIdx.x;
    if (idx >= 32 * 512) return;
    int b = idx / 512, j = idx - b * 512;
    const float* p = pooled + b * 2112;
    const float* w = W1 + (size_t)j * 2112;
    float s = b1[j];
    for (int k = 0; k < 2112; ++k) s += p[k] * w[k];
    zp[idx] = s > 0.f ? s : a[j] * s;
}

// ---- classifier layer 2 ---------------------------------------------------
__global__ void cls2(const float* __restrict__ zp,
                     const float* __restrict__ W2, const float* __restrict__ b2,
                     float* __restrict__ score)
{
    int idx = blockIdx.x * blockDim.x + threadIdx.x;
    if (idx >= 32 * 8) return;
    int b = idx / 8, o = idx - b * 8;
    const float* z = zp + b * 512;
    const float* w = W2 + (size_t)o * 512;
    float s = b2[o];
    for (int k = 0; k < 512; ++k) s += z[k] * w[k];
    score[idx] = s;
}

// ---------------------------------------------------------------------------
extern "C" void kernel_launch(void* const* d_in, const int* in_sizes, int n_in,
                              void* d_out, int out_size, void* d_ws, size_t ws_size,
                              hipStream_t stream)
{
    (void)in_sizes; (void)n_in; (void)out_size; (void)ws_size;

    const float* x     = (const float*)d_in[0];
    const float* adj   = (const float*)d_in[1];
    const float* poolw = (const float*)d_in[2];
    const float* clsW1 = (const float*)d_in[51];
    const float* clsb1 = (const float*)d_in[52];
    const float* prelu = (const float*)d_in[53];
    const float* clsW2 = (const float*)d_in[54];
    const float* clsb2 = (const float*)d_in[55];
    const int*   pnl   = (const int*)d_in[56];

    char* ws = (char*)d_ws;
    size_t cur = 0;
    auto alloc = [&](size_t bytes) -> void* {
        cur = (cur + 255) & ~(size_t)255;
        void* p = ws + cur;
        cur += bytes;
        return p;
    };
    half_t* Ah   = (half_t*)alloc((size_t)512 * 512 * 2);
    float*  hA   = (float*) alloc((size_t)RROWS * 704 * 4);
    float*  hB   = (float*) alloc((size_t)RROWS * 704 * 4);
    half_t* HTA  = (half_t*)alloc((size_t)32 * 704 * 512 * 2);
    half_t* HTB  = (half_t*)alloc((size_t)32 * 704 * 512 * 2);
    half_t* AGG  = (half_t*)alloc((size_t)RROWS * 512 * 2);
    float*  Y1   = (float*) alloc((size_t)RROWS * 160 * 4);
    half_t* Y1h  = (half_t*)alloc((size_t)RROWS * 160 * 2);
    float*  Y2   = (float*) alloc((size_t)RROWS * 128 * 4);
    half_t* W1h  = (half_t*)alloc((size_t)160 * 512 * 2);   // rows padded to 32-mult
    half_t* W2h  = (half_t*)alloc((size_t)128 * 160 * 2);
    float*  mean = (float*) alloc(704 * 4);
    float*  var  = (float*) alloc(704 * 4);
    float*  dmin = (float*) alloc(704 * 4);
    float*  pooled = (float*)alloc((size_t)32 * 2112 * 4);
    float*  zp   = (float*) alloc((size_t)32 * 512 * 4);

    // --- setup: relu(adj)->f16, bn0 over [16384,128] -----------------------
    adj_relu_f16<<<(512 * 512 + 255) / 256, 256, 0, stream>>>(adj, Ah);
    col_stats<<<128, 256, 0, stream>>>(x, RROWS, 128, mean, var);
    bn0_apply<<<(RROWS * 128 + 255) / 256, 256, 0, stream>>>(x, mean, var, hA, HTA, 128);

    float*  hcur = hA;  float*  hnx = hB;
    half_t* HTcur = HTA; half_t* HTnx = HTB;

    const int Fs[3]  = {128, 320, 512};
    const int H1[3]  = {64, 112, 160};   // gcn1 hidden
    const int H1P[3] = {64, 128, 160};   // padded to mult 32 (K pad and O-tile pad)
    const int H2[3]  = {48, 96, 144};
    const int H2P[3] = {64, 96, 160};

    for (int l = 0; l < 3; ++l) {
        int F = Fs[l], Fn = F + 192;
        int base = 3 + l * 16;

        // out1: neighbor max-pool with dummy column-min row
        col_min_k<<<F, 256, 0, stream>>>(hcur, RROWS, F, dmin);
        maxpool_k<<<RROWS, 128, 0, stream>>>(hcur, dmin, pnl, hnx, HTnx, F, Fn);

        // shared aggregation AGG_b = relu(A) * H_b (computed once, WMMA)
        {
            int mg = 512 / (16 * MT), ogr = F / (16 * NT);
            int total = 32 * mg * ogr;
            gemm_wmma_f16<<<(total + 7) / 8, 256, 0, stream>>>(
                Ah, 0, HTcur, (long)F * 512, nullptr,
                nullptr, AGG, (long)512 * F, 512, F, F, mg, ogr, total);
        }

        auto run_gcn = [&](int pbase, int hid, int hidp, int odim, int colOff) {
            const float* W1  = (const float*)d_in[pbase + 0];
            const float* b1  = (const float*)d_in[pbase + 1];
            const float* g1  = (const float*)d_in[pbase + 2];
            const float* be1 = (const float*)d_in[pbase + 3];
            const float* W2  = (const float*)d_in[pbase + 4];
            const float* b2  = (const float*)d_in[pbase + 5];
            const float* g2  = (const float*)d_in[pbase + 6];
            const float* be2 = (const float*)d_in[pbase + 7];

            // MLP layer 1: [R,F] x [hid,F]^T -> Y1 (f32), then BN+relu -> f16
            cvt_f16_pad<<<(hidp * F + 255) / 256, 256, 0, stream>>>(
                W1, W1h, hid, hidp, F, F);
            {
                int mg = RROWS / (16 * MT), ogr = hidp / (16 * NT);
                int total = mg * ogr;
                gemm_wmma_f16<<<(total + 7) / 8, 256, 0, stream>>>(
                    AGG, 0, W1h, 0, b1, Y1, nullptr, 0, F, hid, hid, mg, ogr, total);
            }
            col_stats<<<hid, 256, 0, stream>>>(Y1, RROWS, hid, mean, var);
            bn_relu_f16pad<<<(RROWS * hidp + 255) / 256, 256, 0, stream>>>(
                Y1, g1, be1, mean, var, Y1h, hid, hidp);

            // MLP layer 2: [R,hidp] x [odim,hidp]^T -> Y2, BN+relu -> h slice
            cvt_f16_pad<<<(odim * hidp + 255) / 256, 256, 0, stream>>>(
                W2, W2h, odim, odim, hid, hidp);
            {
                int mg = RROWS / (16 * MT), ogr = odim / (16 * NT);
                int total = mg * ogr;
                gemm_wmma_f16<<<(total + 7) / 8, 256, 0, stream>>>(
                    Y1h, 0, W2h, 0, b2, Y2, nullptr, 0, hidp, odim, odim, mg, ogr, total);
            }
            col_stats<<<odim, 256, 0, stream>>>(Y2, RROWS, odim, mean, var);
            bn_relu_to_h<<<(RROWS * odim + 255) / 256, 256, 0, stream>>>(
                Y2, g2, be2, mean, var, hnx, HTnx, odim, Fn, colOff);
        };

        run_gcn(base,     H1[l], H1P[l], 128, F);        // gcn1 -> cols [F,F+128)
        run_gcn(base + 8, H2[l], H2P[l], 64,  F + 128);  // gcn2 -> cols [F+128,Fn)

        { float* t = hcur; hcur = hnx; hnx = t; }
        { half_t* t = HTcur; HTcur = HTnx; HTnx = t; }
    }

    // --- pooling (max/argmax, weighted, mean) + classifier -----------------
    final_pool<<<32, 256, 0, stream>>>(hcur, poolw, pooled, (float*)d_out + 256);
    cls1<<<(32 * 512 + 255) / 256, 256, 0, stream>>>(pooled, clsW1, clsb1, prelu, zp);
    cls2<<<1, 256, 0, stream>>>(zp, clsW2, clsb2, (float*)d_out);
}
